// OrkyRNNWaaagh_72164040507681
// MI455X (gfx1250) — matrix operand order repaired and checked by reading, not compiled, not run on hardware
//
#include <hip/hip_runtime.h>
#include <hip/hip_bf16.h>
#include <math.h>

typedef __bf16 bf16;
typedef __attribute__((ext_vector_type(8)))  __bf16 v8bf;
typedef __attribute__((ext_vector_type(16))) __bf16 v16bf;
typedef __attribute__((ext_vector_type(8)))  float  v8f;

#define SEQ   256
#define BATCH 8
#define EMB   512
#define HID   1024
#define VOCAB 32000

// ---- WMMA operand loaders (wave32, 16x16x32 bf16) -------------------------
// A (16x32, bf16): lane = m + 16*h ; lane<16 holds K {0..7, 16..23}, lane>=16
// holds K {8..15, 24..31}.  Caller passes p = rowBase + k0 + ((lane>>4)*8).
__device__ __forceinline__ v16bf load_a16(const bf16* p) {
  v8bf lo = *reinterpret_cast<const v8bf*>(p);       // k .. k+7
  v8bf hi = *reinterpret_cast<const v8bf*>(p + 16);  // k+16 .. k+23
  return __builtin_shufflevector(lo, hi, 0,1,2,3,4,5,6,7,8,9,10,11,12,13,14,15);
}
// B (32x16, bf16) column-major source W[n,k]: lane n = lane&15, same K order as A;
// caller passes p = rowBase(n) + k0 + ((lane>>4)*8).
__device__ __forceinline__ v16bf load_b16(const bf16* p) {
  v8bf lo = *reinterpret_cast<const v8bf*>(p);
  v8bf hi = *reinterpret_cast<const v8bf*>(p + 16);
  return __builtin_shufflevector(lo, hi, 0,1,2,3,4,5,6,7,8,9,10,11,12,13,14,15);
}

__device__ __forceinline__ float sigmoidf_(float x) {
  return 1.0f / (1.0f + __expf(-x));
}

#define WMMA_BF16(A, B, C) \
  __builtin_amdgcn_wmma_f32_16x16x32_bf16(false, (A), false, (B), (short)0, (C), false, false)

// ---- conversion / setup kernels -------------------------------------------
__global__ void embed_cvt(const int* __restrict__ tokens,
                          const float* __restrict__ emb,
                          bf16* __restrict__ xbf) {
  int t = blockIdx.x;                 // 0 .. B*S-1  (= b*S + s)
  int tok = tokens[t];
  if (tok < 0 || tok >= VOCAB) tok = 0;
  const float* src = emb + (size_t)tok * EMB;
  bf16* dst = xbf + (size_t)t * EMB;
  for (int e = threadIdx.x; e < EMB; e += blockDim.x)
    dst[e] = (bf16)src[e];
}

__global__ void cvt_gates(const float* __restrict__ Wf, const float* __restrict__ bF,
                          const float* __restrict__ Wi, const float* __restrict__ bI,
                          const float* __restrict__ Wc, const float* __restrict__ bC,
                          const float* __restrict__ Wo, const float* __restrict__ bO,
                          bf16* __restrict__ Wcat, float* __restrict__ bcat, int Kdim) {
  int n = blockIdx.x;                 // 0..4095 : stacked [f|i|c|o] rows
  int g = n >> 10, u = n & 1023;
  const float* W  = (g == 0) ? Wf : (g == 1) ? Wi : (g == 2) ? Wc : Wo;
  const float* bs = (g == 0) ? bF : (g == 1) ? bI : (g == 2) ? bC : bO;
  const float* srow = W + (size_t)u * Kdim;
  bf16* drow = Wcat + (size_t)n * Kdim;
  for (int k = threadIdx.x; k < Kdim; k += blockDim.x)
    drow[k] = (bf16)srow[k];
  if (threadIdx.x == 0) bcat[n] = bs[u];
}

__global__ void cvt_wout(const float* __restrict__ W, bf16* __restrict__ Wb) {
  size_t total = (size_t)VOCAB * HID;
  size_t step  = (size_t)gridDim.x * blockDim.x;
  for (size_t i = (size_t)blockIdx.x * blockDim.x + threadIdx.x; i < total; i += step)
    Wb[i] = (bf16)W[i];
}

__global__ void zero_state(float* c0, float* c1,
                           bf16* h00, bf16* h01, bf16* h10, bf16* h11) {
  int i = blockIdx.x * blockDim.x + threadIdx.x;
  if (i < BATCH * HID) {
    c0[i] = 0.0f; c1[i] = 0.0f;
    h00[i] = (bf16)0.0f; h01[i] = (bf16)0.0f;
    h10[i] = (bf16)0.0f; h11[i] = (bf16)0.0f;
  }
}

// ---- one LSTM layer, one timestep: gates GEMM + pointwise, fused ----------
// A rows (batch) come from two concatenated sources: P0 (k < split), P1 (rest).
// Wcat is [4096, Ktot] bf16 = stacked [f|i|c|o] gate weights, W[n,k] layout.
// Software-pipelined: loads for K-block k+32 are issued before the WMMAs of k.
__global__ __launch_bounds__(128) void lstm_step(
    const bf16* __restrict__ P0, long stride0,
    const bf16* __restrict__ P1, long stride1,
    int split, int Ktot,
    const bf16* __restrict__ Wcat, const float* __restrict__ bcat,
    float* __restrict__ cstate,
    bf16* __restrict__ hnew,
    bf16* __restrict__ hseq, long hseq_stride) {
  int lane = threadIdx.x & 31;
  int wid  = blockIdx.x * (blockDim.x >> 5) + (threadIdx.x >> 5);  // 0..63
  int u0   = wid << 4;          // 16-wide hidden-unit tile
  int row  = lane & 15;
  int mrow = row & 7;           // rows 8..15 alias 0..7 (outputs discarded)
  const bf16* a0 = P0 + (long)mrow * stride0;
  const bf16* a1 = P1 + (long)mrow * stride1;
  int kA = (lane >> 4) << 3;
  int kB = (lane >> 4) << 3;
  const bf16* bFp = Wcat + ((size_t)(0 * HID + u0 + row)) * Ktot + kB;
  const bf16* bIp = Wcat + ((size_t)(1 * HID + u0 + row)) * Ktot + kB;
  const bf16* bCp = Wcat + ((size_t)(2 * HID + u0 + row)) * Ktot + kB;
  const bf16* bOp = Wcat + ((size_t)(3 * HID + u0 + row)) * Ktot + kB;

  v8f accF = {}, accI = {}, accG = {}, accO = {};

  // prologue: K-block 0 (always in the P0 region; split >= 512)
  v16bf a  = load_a16(a0 + kA);
  v16bf wf = load_b16(bFp);
  v16bf wi = load_b16(bIp);
  v16bf wc = load_b16(bCp);
  v16bf wo = load_b16(bOp);

#pragma unroll 2
  for (int k0 = 32; k0 < Ktot; k0 += 32) {
    const bf16* an = (k0 < split) ? (a0 + k0) : (a1 + (k0 - split));
    v16bf a2  = load_a16(an + kA);
    v16bf wf2 = load_b16(bFp + k0);
    v16bf wi2 = load_b16(bIp + k0);
    v16bf wc2 = load_b16(bCp + k0);
    v16bf wo2 = load_b16(bOp + k0);
    accF = WMMA_BF16(a, wf, accF);
    accI = WMMA_BF16(a, wi, accI);
    accG = WMMA_BF16(a, wc, accG);
    accO = WMMA_BF16(a, wo, accO);
    a = a2; wf = wf2; wi = wi2; wc = wc2; wo = wo2;
  }
  // epilogue
  accF = WMMA_BF16(a, wf, accF);
  accI = WMMA_BF16(a, wi, accI);
  accG = WMMA_BF16(a, wc, accG);
  accO = WMMA_BF16(a, wo, accO);

  // D layout: lanes 0..15 hold M=0..7 in VGPRs 0..7, N = lane.  Only those are valid.
  if (lane < 16) {
    int u = u0 + lane;
    float biasF = bcat[u];
    float biasI = bcat[HID + u];
    float biasC = bcat[2 * HID + u];
    float biasO = bcat[3 * HID + u];
#pragma unroll
    for (int r = 0; r < 8; ++r) {
      float fg = sigmoidf_(accF[r] + biasF);
      float ig = sigmoidf_(accI[r] + biasI);
      float gg = tanhf(accG[r] + biasC);
      float og = sigmoidf_(accO[r] + biasO);
      size_t ci = (size_t)r * HID + u;
      float cn = fg * cstate[ci] + ig * gg;
      cstate[ci] = cn;
      float hv = og * tanhf(cn);
      hnew[ci] = (bf16)hv;
      if (hseq) hseq[(long)r * hseq_stride + u] = (bf16)hv;
    }
  }
}

// ---- output projection: [2048 x 1024] x [1024 x 32000] --------------------
// Each wave: one 16x64 tile (4 WMMA accumulators sharing the A load).
// Software-pipelined + speculative prefetch of the streamed Wout rows.
__global__ __launch_bounds__(256) void proj_kernel(
    const bf16* __restrict__ Hbf, const bf16* __restrict__ Wb,
    const float* __restrict__ bout, float* __restrict__ out) {
  const int NSTRIP = VOCAB / 64;      // 500 strips of 64 columns
  int lane = threadIdx.x & 31;
  int wid  = blockIdx.x * (blockDim.x >> 5) + (threadIdx.x >> 5);
  int mtile = wid / NSTRIP;
  int strip = wid - mtile * NSTRIP;
  if (mtile >= (BATCH * SEQ) / 16) return;
  int m0 = mtile << 4;
  int n0 = strip << 6;
  int row = lane & 15;
  int kA = (lane >> 4) << 3;
  int kB = (lane >> 4) << 3;
  const bf16* arow = Hbf + (size_t)(m0 + row) * HID + kA;
  const bf16* b0 = Wb + (size_t)(n0 +  0 + row) * HID + kB;
  const bf16* b1 = Wb + (size_t)(n0 + 16 + row) * HID + kB;
  const bf16* b2 = Wb + (size_t)(n0 + 32 + row) * HID + kB;
  const bf16* b3 = Wb + (size_t)(n0 + 48 + row) * HID + kB;

  v8f acc0 = {}, acc1 = {}, acc2 = {}, acc3 = {};

  // prologue: K-block 0
  v16bf a  = load_a16(arow);
  v16bf w0 = load_b16(b0);
  v16bf w1 = load_b16(b1);
  v16bf w2 = load_b16(b2);
  v16bf w3 = load_b16(b3);

#pragma unroll 2
  for (int k0 = 32; k0 < HID; k0 += 32) {
    // speculative prefetch of the weight stream, 8 K-blocks (512B/row) ahead
    if ((k0 & 255) == 32) {
      __builtin_prefetch((const void*)(b0 + k0 + 256), 0, 1);
      __builtin_prefetch((const void*)(b1 + k0 + 256), 0, 1);
      __builtin_prefetch((const void*)(b2 + k0 + 256), 0, 1);
      __builtin_prefetch((const void*)(b3 + k0 + 256), 0, 1);
    }
    v16bf a2  = load_a16(arow + k0);
    v16bf w02 = load_b16(b0 + k0);
    v16bf w12 = load_b16(b1 + k0);
    v16bf w22 = load_b16(b2 + k0);
    v16bf w32 = load_b16(b3 + k0);
    acc0 = WMMA_BF16(a, w0, acc0);
    acc1 = WMMA_BF16(a, w1, acc1);
    acc2 = WMMA_BF16(a, w2, acc2);
    acc3 = WMMA_BF16(a, w3, acc3);
    a = a2; w0 = w02; w1 = w12; w2 = w22; w3 = w32;
  }
  // epilogue
  acc0 = WMMA_BF16(a, w0, acc0);
  acc1 = WMMA_BF16(a, w1, acc1);
  acc2 = WMMA_BF16(a, w2, acc2);
  acc3 = WMMA_BF16(a, w3, acc3);

  int mhi = (lane >> 4) << 3;         // lanes 16..31 hold rows M=8..15
  int col = lane & 15;
#pragma unroll
  for (int r = 0; r < 8; ++r) {
    size_t m = (size_t)(m0 + mhi + r);
    float* orow = out + m * VOCAB + n0 + col;
    orow[0]  = acc0[r] + bout[n0 + col];
    orow[16] = acc1[r] + bout[n0 + 16 + col];
    orow[32] = acc2[r] + bout[n0 + 32 + col];
    orow[48] = acc3[r] + bout[n0 + 48 + col];
  }
}

// ---- host orchestration ----------------------------------------------------
extern "C" void kernel_launch(void* const* d_in, const int* in_sizes, int n_in,
                              void* d_out, int out_size, void* d_ws, size_t ws_size,
                              hipStream_t stream) {
  (void)in_sizes; (void)n_in; (void)out_size; (void)ws_size;
  const int*   tokens = (const int*)d_in[0];
  const float* emb    = (const float*)d_in[1];
  const float* Wf0 = (const float*)d_in[2],  *bf0 = (const float*)d_in[3];
  const float* Wi0 = (const float*)d_in[4],  *bi0 = (const float*)d_in[5];
  const float* Wc0 = (const float*)d_in[6],  *bc0 = (const float*)d_in[7];
  const float* Wo0 = (const float*)d_in[8],  *bo0 = (const float*)d_in[9];
  const float* Wf1 = (const float*)d_in[10], *bf1 = (const float*)d_in[11];
  const float* Wi1 = (const float*)d_in[12], *bi1 = (const float*)d_in[13];
  const float* Wc1 = (const float*)d_in[14], *bc1 = (const float*)d_in[15];
  const float* Wo1 = (const float*)d_in[16], *bo1 = (const float*)d_in[17];
  const float* Wout = (const float*)d_in[18], *bout = (const float*)d_in[19];
  float* out = (float*)d_out;

  char* ws = (char*)d_ws;
  size_t off = 0;
  auto alloc = [&](size_t bytes) -> char* {
    char* p = ws + off;
    off = (off + bytes + 255) & ~(size_t)255;
    return p;
  };
  const int K0 = EMB + HID;     // 1536
  const int K1 = 2 * HID;       // 2048
  bf16*  Wcat0 = (bf16*)alloc((size_t)4 * HID * K0 * sizeof(bf16));
  bf16*  Wcat1 = (bf16*)alloc((size_t)4 * HID * K1 * sizeof(bf16));
  bf16*  Woutb = (bf16*)alloc((size_t)VOCAB * HID * sizeof(bf16));
  float* bcat0 = (float*)alloc((size_t)4 * HID * sizeof(float));
  float* bcat1 = (float*)alloc((size_t)4 * HID * sizeof(float));
  bf16*  xbf   = (bf16*)alloc((size_t)BATCH * SEQ * EMB * sizeof(bf16));
  bf16*  h0bufA = (bf16*)alloc((size_t)BATCH * HID * sizeof(bf16));
  bf16*  h0bufB = (bf16*)alloc((size_t)BATCH * HID * sizeof(bf16));
  bf16*  h1bufA = (bf16*)alloc((size_t)BATCH * HID * sizeof(bf16));
  bf16*  h1bufB = (bf16*)alloc((size_t)BATCH * HID * sizeof(bf16));
  float* c0    = (float*)alloc((size_t)BATCH * HID * sizeof(float));
  float* c1    = (float*)alloc((size_t)BATCH * HID * sizeof(float));
  bf16*  hsbf  = (bf16*)alloc((size_t)BATCH * SEQ * HID * sizeof(bf16));
  bf16* h0buf[2] = {h0bufA, h0bufB};
  bf16* h1buf[2] = {h1bufA, h1bufB};

  cvt_gates<<<4 * HID, 256, 0, stream>>>(Wf0, bf0, Wi0, bi0, Wc0, bc0, Wo0, bo0, Wcat0, bcat0, K0);
  cvt_gates<<<4 * HID, 256, 0, stream>>>(Wf1, bf1, Wi1, bi1, Wc1, bc1, Wo1, bo1, Wcat1, bcat1, K1);
  cvt_wout<<<16384, 256, 0, stream>>>(Wout, Woutb);
  embed_cvt<<<BATCH * SEQ, 256, 0, stream>>>(tokens, emb, xbf);
  zero_state<<<(BATCH * HID + 255) / 256, 256, 0, stream>>>(c0, c1, h0buf[0], h0buf[1], h1buf[0], h1buf[1]);

  for (int s = 0; s < SEQ; ++s) {
    const bf16* h0p = h0buf[s & 1];
    bf16*       h0n = h0buf[(s + 1) & 1];
    const bf16* h1p = h1buf[s & 1];
    bf16*       h1n = h1buf[(s + 1) & 1];
    // layer 0: A = [ x_t | h0_{t-1} ], K = 1536
    lstm_step<<<16, 128, 0, stream>>>(xbf + (size_t)s * EMB, (long)SEQ * EMB,
                                      h0p, (long)HID, EMB, K0,
                                      Wcat0, bcat0, c0, h0n,
                                      (bf16*)nullptr, 0L);
    // layer 1: A = [ h0_t | h1_{t-1} ], K = 2048; also scatter h1 into hs_seq
    lstm_step<<<16, 128, 0, stream>>>(h0n, (long)HID,
                                      h1p, (long)HID, HID, K1,
                                      Wcat1, bcat1, c1, h1n,
                                      hsbf + (size_t)s * HID, (long)SEQ * HID);
  }

  // logits = hs_seq @ Wout.T + bout : 64000 waves exactly
  proj_kernel<<<8000, 256, 0, stream>>>(hsbf, Woutb, bout, out);
}
